// TwoStageSelfgatingRNN_61040075210839
// MI455X (gfx1250) — compile-verified, matrix operand order checked by reading, not run on hardware
//
#include <hip/hip_runtime.h>

// ---------------------------------------------------------------------------
// TwoStageSelfgatingRNN for MI455X (gfx1250, wave32, WMMA)
//   V = X@Wv ; K = exp(X@Wk) ; Q = sigmoid(X@Wq)
//   S = cumsum(K*V,axis=L) / (cumsum(K,axis=L)+eps) ; out = (S*Q)@Wc
// f16 inputs + f32 WMMA accumulation (v_wmma_f32_16x16x32_f16) for all four
// GEMMs; async global->LDS staging (ASYNCcnt) with double-buffered LDS;
// chunked two-pass parallel scan for the running-sum gating.
// ---------------------------------------------------------------------------

typedef __attribute__((ext_vector_type(16))) _Float16 v16h;
typedef __attribute__((ext_vector_type(8)))  _Float16 v8h;
typedef __attribute__((ext_vector_type(8)))  float    v8f;
typedef __attribute__((ext_vector_type(4)))  int      v4i;

constexpr int   DIM     = 1024;
constexpr int   BB      = 4;
constexpr int   LL      = 8192;
constexpr int   MTOT    = BB * LL;        // 32768 rows
constexpr float EPSV    = 1e-6f;

constexpr int BM  = 64;                   // block tile rows
constexpr int BN  = 64;                   // block tile cols
constexpr int BK  = 32;                   // k step (WMMA K)
constexpr int LDT = 40;                   // padded LDS stride (halves): 80B rows

constexpr int NCHUNK = 64;                // scan chunks along L
constexpr int CHUNK  = LL / NCHUNK;       // 128

// workspace layout (bytes)
constexpr size_t SZ_XH  = (size_t)MTOT * DIM * 2;   //  64 MiB f16 X
constexpr size_t SZ_W   = (size_t)DIM * DIM * 2;    //   2 MiB f16 weight
constexpr size_t SZ_F32 = (size_t)MTOT * DIM * 4;   // 128 MiB f32 plane
constexpr size_t OFF_WQ = SZ_XH;
constexpr size_t OFF_WK = OFF_WQ + SZ_W;
constexpr size_t OFF_WV = OFF_WK + SZ_W;
constexpr size_t OFF_WC = OFF_WV + SZ_W;
constexpr size_t OFF_K  = OFF_WC + SZ_W;
constexpr size_t OFF_Q  = OFF_K  + SZ_F32;          // f16 Q plane
constexpr size_t OFF_O  = OFF_Q  + SZ_XH;           // f16 O plane
constexpr size_t OFF_PK = OFF_O  + SZ_XH;
constexpr size_t OFF_PKV= OFF_PK + (size_t)BB * NCHUNK * DIM * 4;

// ----------------------------------------------------- async global -> LDS
__device__ __forceinline__ void async_cp_b128(_Float16* lds, const _Float16* g) {
#if __has_builtin(__builtin_amdgcn_global_load_async_to_lds_b128)
  __builtin_amdgcn_global_load_async_to_lds_b128(
      (__attribute__((address_space(1))) v4i*)(g),
      (__attribute__((address_space(3))) v4i*)(lds), 0, 0);
#else
  *(v8h*)lds = *(const v8h*)g;
#endif
}

template <int N>
__device__ __forceinline__ void async_wait() {
#if __has_builtin(__builtin_amdgcn_s_wait_asynccnt)
  __builtin_amdgcn_s_wait_asynccnt(N);
#endif
}

// -------------------------------------------------------------------- utils
__device__ __forceinline__ v16h frag_ld(const _Float16* p, int g) {
  // WMMA 16-bit fragment: two contiguous 16B k-chunks per lane
  v8h lo = *(const v8h*)(p + 8 * g);
  v8h hi = *(const v8h*)(p + 16 + 8 * g);
  return __builtin_shufflevector(lo, hi, 0,1,2,3,4,5,6,7,8,9,10,11,12,13,14,15);
}

// ----------------------------------------------------------- convert X->f16
__global__ void convert_x(const float* __restrict__ X, _Float16* __restrict__ Xh,
                          int n) {
  int i = blockIdx.x * blockDim.x + threadIdx.x;
  int stride = gridDim.x * blockDim.x;
  for (; i < n; i += stride) Xh[i] = (_Float16)X[i];
}

// -------------------------------------------- transpose+convert W -> WT f16
__global__ void transpose_cvt(const float* __restrict__ W,
                              _Float16* __restrict__ WT) {
  __shared__ _Float16 tile[32][33];
  const int bx = blockIdx.x % (DIM / 32);
  const int by = blockIdx.x / (DIM / 32);
  const int tx = threadIdx.x % 32;
  const int ty = threadIdx.x / 32;   // 0..7
  #pragma unroll
  for (int i = 0; i < 32; i += 8) {
    tile[ty + i][tx] = (_Float16)W[(size_t)(by * 32 + ty + i) * DIM + bx * 32 + tx];
  }
  __syncthreads();
  #pragma unroll
  for (int i = 0; i < 32; i += 8) {
    WT[(size_t)(bx * 32 + ty + i) * DIM + by * 32 + tx] = tile[tx][ty + i];
  }
}

// --------------------------------------------------- fused 3-way projection
//   accV = Xh@Wv ; accK = Xh@Wk ; accQ = Xh@Wq  -> K=exp, KV=K*V, Q=sigmoid
__global__ __launch_bounds__(128) void gemm3_kernel(
    const _Float16* __restrict__ A,
    const _Float16* __restrict__ BvT, const _Float16* __restrict__ BkT,
    const _Float16* __restrict__ BqT,
    float* __restrict__ Kout, float* __restrict__ KVout,
    _Float16* __restrict__ Qout) {
  __shared__ __align__(16) _Float16 As[2][BM * LDT];
  __shared__ __align__(16) _Float16 Bs[2][3][BN * LDT];

  const int tid  = threadIdx.x;
  const int lane = tid & 31;
  const int wave = tid >> 5;
  const int g    = lane >> 4;
  const int mn   = lane & 15;
  const int nt   = DIM / BN;                 // 16
  const int row0 = (blockIdx.x / nt) * BM;
  const int col0 = (blockIdx.x % nt) * BN;
  const int wm0  = (wave >> 1) * 32;
  const int wn0  = (wave & 1) * 32;

  const _Float16* Bt[3] = {BvT, BkT, BqT};
  v8f acc[3][2][2] = {};

  // stage one k-slab (8 async b128 per thread) into buffer `buf`
  auto stage = [&](int buf, int k0) {
    #pragma unroll
    for (int i = 0; i < 2; ++i) {
      const int c = tid * 2 + i, r = c >> 2, kc = (c & 3) * 8;
      async_cp_b128(As[buf] + r * LDT + kc,
                    A + (size_t)(row0 + r) * DIM + k0 + kc);
    }
    #pragma unroll
    for (int w = 0; w < 3; ++w) {
      #pragma unroll
      for (int i = 0; i < 2; ++i) {
        const int c = tid * 2 + i, r = c >> 2, kc = (c & 3) * 8;
        async_cp_b128(Bs[buf][w] + r * LDT + kc,
                      Bt[w] + (size_t)(col0 + r) * DIM + k0 + kc);
      }
    }
  };

  stage(0, 0);
  for (int k0 = 0, it = 0; k0 < DIM; k0 += BK, ++it) {
    const int buf = it & 1;
    if (k0 + BK < DIM) {
      stage(buf ^ 1, k0 + BK);  // overlap next slab with this slab's WMMAs
      async_wait<8>();          // in-order: current slab (8 older ops) done
    } else {
      async_wait<0>();
    }
    __syncthreads();

    v16h af[2];
    #pragma unroll
    for (int t = 0; t < 2; ++t)
      af[t] = frag_ld(As[buf] + (wm0 + t * 16 + mn) * LDT, g);

    #pragma unroll
    for (int w = 0; w < 3; ++w) {
      v16h bf[2];
      #pragma unroll
      for (int t = 0; t < 2; ++t)
        bf[t] = frag_ld(Bs[buf][w] + (wn0 + t * 16 + mn) * LDT, g);
      #pragma unroll
      for (int i = 0; i < 2; ++i)
        #pragma unroll
        for (int j = 0; j < 2; ++j)
          acc[w][i][j] = __builtin_amdgcn_wmma_f32_16x16x32_f16(
              false, af[i], false, bf[j], (short)0, acc[w][i][j], false, false);
    }
    __syncthreads();  // all waves done reading buf before it is restaged
  }

  #pragma unroll
  for (int i = 0; i < 2; ++i) {
    #pragma unroll
    for (int j = 0; j < 2; ++j) {
      const int colg = col0 + wn0 + j * 16 + mn;
      #pragma unroll
      for (int r = 0; r < 8; ++r) {
        const int rowg  = row0 + wm0 + i * 16 + r + 8 * g;
        const size_t off = (size_t)rowg * DIM + colg;
        const float v  = acc[0][i][j][r];
        const float kk = __expf(acc[1][i][j][r]);
        const float qq = 1.0f / (1.0f + __expf(-acc[2][i][j][r]));
        Kout[off]  = kk;
        KVout[off] = kk * v;
        Qout[off]  = (_Float16)qq;
      }
    }
  }
}

// ------------------------------------------------- chunked scan: pass A sums
__global__ void scan_partials(const float* __restrict__ Kin,
                              const float* __restrict__ KVin,
                              float* __restrict__ PK, float* __restrict__ PKV) {
  const int idx = blockIdx.x * blockDim.x + threadIdx.x;   // (b,c,d)
  const int d = idx % DIM;
  const int c = (idx / DIM) % NCHUNK;
  const int b = idx / (DIM * NCHUNK);
  const size_t base = ((size_t)b * LL + (size_t)c * CHUNK) * DIM + d;
  float sk = 0.f, skv = 0.f;
  #pragma unroll 4
  for (int l = 0; l < CHUNK; ++l) {
    const size_t off = base + (size_t)l * DIM;
    sk += Kin[off];
    skv += KVin[off];
  }
  PK[idx]  = sk;
  PKV[idx] = skv;
}

// --------------------------------------- pass B: prefix + gate, O = S*Q f16
__global__ void scan_apply(const float* __restrict__ Kin,
                           const float* __restrict__ KVin,
                           const _Float16* __restrict__ Qin,
                           const float* __restrict__ PK,
                           const float* __restrict__ PKV,
                           _Float16* __restrict__ Oout) {
  const int idx = blockIdx.x * blockDim.x + threadIdx.x;
  const int d = idx % DIM;
  const int c = (idx / DIM) % NCHUNK;
  const int b = idx / (DIM * NCHUNK);
  float sk = 0.f, skv = 0.f;
  const size_t pbase = (size_t)b * NCHUNK * DIM + d;
  for (int cc = 0; cc < c; ++cc) {
    sk  += PK[pbase + (size_t)cc * DIM];
    skv += PKV[pbase + (size_t)cc * DIM];
  }
  const size_t base = ((size_t)b * LL + (size_t)c * CHUNK) * DIM + d;
  #pragma unroll 4
  for (int l = 0; l < CHUNK; ++l) {
    const size_t off = base + (size_t)l * DIM;
    sk  += Kin[off];
    skv += KVin[off];
    const float s = skv / (sk + EPSV);
    Oout[off] = (_Float16)(s * (float)Qin[off]);
  }
}

// --------------------------------------------------- final GEMM: O @ Wc f32
__global__ __launch_bounds__(128) void gemmc_kernel(
    const _Float16* __restrict__ A, const _Float16* __restrict__ BcT,
    float* __restrict__ Cout) {
  __shared__ __align__(16) _Float16 As[2][BM * LDT];
  __shared__ __align__(16) _Float16 Bs[2][BN * LDT];

  const int tid  = threadIdx.x;
  const int lane = tid & 31;
  const int wave = tid >> 5;
  const int g    = lane >> 4;
  const int mn   = lane & 15;
  const int nt   = DIM / BN;
  const int row0 = (blockIdx.x / nt) * BM;
  const int col0 = (blockIdx.x % nt) * BN;
  const int wm0  = (wave >> 1) * 32;
  const int wn0  = (wave & 1) * 32;

  v8f acc[2][2] = {};

  auto stage = [&](int buf, int k0) {
    #pragma unroll
    for (int i = 0; i < 2; ++i) {
      const int c = tid * 2 + i, r = c >> 2, kc = (c & 3) * 8;
      async_cp_b128(As[buf] + r * LDT + kc,
                    A + (size_t)(row0 + r) * DIM + k0 + kc);
      async_cp_b128(Bs[buf] + r * LDT + kc,
                    BcT + (size_t)(col0 + r) * DIM + k0 + kc);
    }
  };

  stage(0, 0);
  for (int k0 = 0, it = 0; k0 < DIM; k0 += BK, ++it) {
    const int buf = it & 1;
    if (k0 + BK < DIM) {
      stage(buf ^ 1, k0 + BK);
      async_wait<4>();
    } else {
      async_wait<0>();
    }
    __syncthreads();

    v16h af[2], bf[2];
    #pragma unroll
    for (int t = 0; t < 2; ++t) {
      af[t] = frag_ld(As[buf] + (wm0 + t * 16 + mn) * LDT, g);
      bf[t] = frag_ld(Bs[buf] + (wn0 + t * 16 + mn) * LDT, g);
    }
    #pragma unroll
    for (int i = 0; i < 2; ++i)
      #pragma unroll
      for (int j = 0; j < 2; ++j)
        acc[i][j] = __builtin_amdgcn_wmma_f32_16x16x32_f16(
            false, af[i], false, bf[j], (short)0, acc[i][j], false, false);
    __syncthreads();
  }

  #pragma unroll
  for (int i = 0; i < 2; ++i) {
    #pragma unroll
    for (int j = 0; j < 2; ++j) {
      const int colg = col0 + wn0 + j * 16 + mn;
      #pragma unroll
      for (int r = 0; r < 8; ++r) {
        const int rowg = row0 + wm0 + i * 16 + r + 8 * g;
        Cout[(size_t)rowg * DIM + colg] = acc[i][j][r];
      }
    }
  }
}

// --------------------------------------------------------------------- host
extern "C" void kernel_launch(void* const* d_in, const int* in_sizes, int n_in,
                              void* d_out, int out_size, void* d_ws,
                              size_t ws_size, hipStream_t stream) {
  (void)in_sizes; (void)n_in; (void)out_size; (void)ws_size;
  const float* X  = (const float*)d_in[0];
  const float* Wq = (const float*)d_in[1];
  const float* Wk = (const float*)d_in[2];
  const float* Wv = (const float*)d_in[3];
  const float* Wc = (const float*)d_in[4];
  float* out = (float*)d_out;
  char*  ws  = (char*)d_ws;

  _Float16* Xh  = (_Float16*)(ws);
  _Float16* WqT = (_Float16*)(ws + OFF_WQ);
  _Float16* WkT = (_Float16*)(ws + OFF_WK);
  _Float16* WvT = (_Float16*)(ws + OFF_WV);
  _Float16* WcT = (_Float16*)(ws + OFF_WC);
  float*    Kb  = (float*)(ws + OFF_K);
  _Float16* Qb  = (_Float16*)(ws + OFF_Q);
  _Float16* Ob  = (_Float16*)(ws + OFF_O);
  float*    PK  = (float*)(ws + OFF_PK);
  float*    PKV = (float*)(ws + OFF_PKV);
  float*    KVb = out;  // f32 scratch; fully overwritten by gemmc_kernel

  convert_x<<<8192, 256, 0, stream>>>(X, Xh, MTOT * DIM);
  const int tgrid = (DIM / 32) * (DIM / 32);
  transpose_cvt<<<tgrid, 256, 0, stream>>>(Wq, WqT);
  transpose_cvt<<<tgrid, 256, 0, stream>>>(Wk, WkT);
  transpose_cvt<<<tgrid, 256, 0, stream>>>(Wv, WvT);
  transpose_cvt<<<tgrid, 256, 0, stream>>>(Wc, WcT);

  const int gemm_grid = (MTOT / BM) * (DIM / BN);   // 512*16 = 8192
  gemm3_kernel<<<gemm_grid, 128, 0, stream>>>(Xh, WvT, WkT, WqT, Kb, KVb, Qb);

  const int scan_grid = (BB * NCHUNK * DIM) / 256;  // 1024
  scan_partials<<<scan_grid, 256, 0, stream>>>(Kb, KVb, PK, PKV);
  scan_apply<<<scan_grid, 256, 0, stream>>>(Kb, KVb, Qb, PK, PKV, Ob);

  gemmc_kernel<<<gemm_grid, 128, 0, stream>>>(Ob, WcT, out);
}
